// BasicConvClassifierWithSubject_73160472920174
// MI455X (gfx1250) — matrix operand order repaired and checked by reading, not compiled, not run on hardware
//
#include <hip/hip_runtime.h>
#include <hip/hip_bf16.h>
#include <math.h>

typedef __bf16 bf16_t;
typedef __attribute__((ext_vector_type(16))) __bf16 bf16x16;
typedef __attribute__((ext_vector_type(8)))  __bf16 bf16x8;
typedef __attribute__((ext_vector_type(8)))  float  f32x8;

// Problem constants
#define NB    1024
#define CIN0  271
#define TV    281      // valid time steps
#define TP    288      // padded time (mult of 32)
#define CX    288      // padded stem input channels (271 -> 288, mult of 32)
#define H1    128
#define H2    256
#define NE    16
#define NCLS  1854
#define EPS_  1e-5f

__device__ __forceinline__ float gelu_exact(float x) {
    return 0.5f * x * (1.0f + erff(x * 0.70710678118654752f));
}

// ---------------------------------------------------------------------------
// Small prep kernels (front-end algebraic fusion)
// ---------------------------------------------------------------------------

// Row softmax of attention [271x271]
__global__ void k_softmax(const float* __restrict__ A, float* __restrict__ P) {
    const int i = blockIdx.x;
    const int tid = threadIdx.x;            // 128 threads
    __shared__ float red[128];
    float m = -3.4e38f;
    for (int c = tid; c < CIN0; c += 128) m = fmaxf(m, A[i * CIN0 + c]);
    red[tid] = m; __syncthreads();
    for (int s = 64; s > 0; s >>= 1) { if (tid < s) red[tid] = fmaxf(red[tid], red[tid + s]); __syncthreads(); }
    m = red[0]; __syncthreads();
    float sum = 0.0f;
    for (int c = tid; c < CIN0; c += 128) {
        float e = expf(A[i * CIN0 + c] - m);
        P[i * CIN0 + c] = e;
        sum += e;
    }
    red[tid] = sum; __syncthreads();
    for (int s = 64; s > 0; s >>= 1) { if (tid < s) red[tid] += red[tid + s]; __syncthreads(); }
    const float inv = 1.0f / red[0];
    for (int c = tid; c < CIN0; c += 128) P[i * CIN0 + c] *= inv;
}

// W1[o][c] = sum_i sa_w[o][i] * attn[i][c]   (128x271)
__global__ void k_w1(const float* __restrict__ sa_w, const float* __restrict__ attn,
                     float* __restrict__ W1) {
    int idx = blockIdx.x * blockDim.x + threadIdx.x;
    if (idx >= H1 * CIN0) return;
    int o = idx / CIN0, c = idx % CIN0;
    float s = 0.0f;
    for (int i = 0; i < CIN0; ++i) s += sa_w[o * CIN0 + i] * attn[i * CIN0 + c];
    W1[idx] = s;
}

// M[s][o][c] = sum_i subj_w[s][o][i] * W1[i][c], bf16, channel-padded to CX
__global__ void k_msub(const float* __restrict__ subj_w, const float* __restrict__ W1,
                       bf16_t* __restrict__ M) {
    int idx = blockIdx.x * blockDim.x + threadIdx.x;
    if (idx >= 4 * H1 * CX) return;
    int c = idx % CX;
    int o = (idx / CX) % H1;
    int s = idx / (CX * H1);
    float acc = 0.0f;
    if (c < CIN0)
        for (int i = 0; i < H1; ++i)
            acc += subj_w[((size_t)s * H1 + o) * H1 + i] * W1[i * CIN0 + c];
    M[idx] = (bf16_t)acc;
}

// Mb[s][o] = sum_i subj_w[s][o][i] * sa_b[i] + subj_b[s][o]
__global__ void k_mb(const float* __restrict__ subj_w, const float* __restrict__ sa_b,
                     const float* __restrict__ subj_b, float* __restrict__ Mb) {
    int idx = blockIdx.x * blockDim.x + threadIdx.x;
    if (idx >= 4 * H1) return;
    int o = idx % H1, s = idx / H1;
    float a = subj_b[s * H1 + o];
    for (int i = 0; i < H1; ++i) a += subj_w[((size_t)s * H1 + o) * H1 + i] * sa_b[i];
    Mb[idx] = a;
}

// Transpose X [B,271,281] f32 -> Xt [B, TP, CX] bf16 (zero-padded)
__global__ void k_convert_x(const float* __restrict__ X, bf16_t* __restrict__ Xt) {
    const int t = blockIdx.x;      // 0..287
    const int b = blockIdx.y;      // batch
    const int c = threadIdx.x;     // 0..287 (blockDim = 288)
    float v = 0.0f;
    if (t < TV && c < CIN0)
        v = X[((size_t)b * CIN0 + c) * TV + t];
    Xt[((size_t)b * TP + t) * CX + c] = (bf16_t)v;
}

// w [O][I][K] f32  ->  wT [K][O][I] bf16
__global__ void k_wprep(const float* __restrict__ w, bf16_t* __restrict__ wT,
                        int O, int I, int K) {
    int idx = blockIdx.x * blockDim.x + threadIdx.x;
    if (idx >= K * O * I) return;
    int c = idx % I;
    int o = (idx / I) % O;
    int k = idx / (I * O);
    wT[idx] = (bf16_t)w[((size_t)o * I + c) * K + k];
}

// Fold conv bias + eval-mode BN: scale = gamma/sqrt(1+eps); shift = bias*scale + beta
__global__ void k_bnprep(const float* __restrict__ cb, const float* __restrict__ g,
                         const float* __restrict__ be, float* __restrict__ s,
                         float* __restrict__ sh, int n) {
    int i = blockIdx.x * blockDim.x + threadIdx.x;
    if (i >= n) return;
    float sv = g[i] * rsqrtf(1.0f + EPS_);
    s[i] = sv;
    sh[i] = cb[i] * sv + be[i];
}

// ---------------------------------------------------------------------------
// Conv1d-as-GEMM WMMA kernel, LDS-staged via async copy.
//   act  : bf16 [B][TP][CIN]   (time rows >= TV are zero)
//   wT   : bf16 [KTAPS][Cout][CIN]
//   out  : bf16 [B][TP][Cout]
// Block = 4 waves, one 32-time-step super-tile. Activation rows
// [t0-pad, t0+32+pad) are DMA'd into LDS once (global_load_async_to_lds_b128,
// ASYNCcnt), then each wave computes a 16(Cout) x 32(time) tile: two
// v_wmma_f32_16x16x32_bf16 per (tap, 32-chan chunk), sharing the A fragment.
// Epilogue: y = acc*scale + shift (+resid) (gelu?); pad time rows forced to 0.
// subj != null: per-batch weight/bias gather (stem per-subject matrix).
// ---------------------------------------------------------------------------
template<int KTAPS, int CIN>
__global__ __launch_bounds__(128) void conv_wmma(
    const bf16_t* __restrict__ act,
    const bf16_t* __restrict__ wT, int Cout,
    const float* __restrict__ scale,
    const float* __restrict__ shift,
    const bf16_t* __restrict__ resid,
    bf16_t* __restrict__ out,
    int doGelu,
    const int* __restrict__ subj, long wSubjStride, long shSubjStride)
{
    constexpr int NR    = 32 + KTAPS - 1;   // staged rows (incl. halo)
    constexpr int PITCH = CIN + 8;          // +16B skew to spread LDS banks
    constexpr int VPR   = CIN / 8;          // 16B vectors per row
    __shared__ bf16_t sh[NR * PITCH];

    const int tid   = threadIdx.x;
    const int lane  = tid & 31;
    const int wv    = tid >> 5;
    const int n     = lane & 15;            // A-row / D-column index
    const int hh    = lane >> 4;            // half-wave select
    const int t0    = blockIdx.x * 32;
    const int o0    = (blockIdx.y * 4 + wv) * 16;
    const int b     = blockIdx.z;
    const int tbase = t0 - (KTAPS >> 1);

    const bf16_t* actB = act + (size_t)b * TP * CIN;

    // ---- stage activation rows into LDS (async global->LDS DMA) ----
    for (int v = tid; v < NR * VPR; v += 128) {
        const int row = v / VPR;
        const int cv  = v - row * VPR;
        const int t   = tbase + row;
        bf16_t* dstp = &sh[row * PITCH + cv * 8];
        if ((unsigned)t < (unsigned)TV) {
            const unsigned ldsOff = (unsigned)(uintptr_t)dstp;     // low 32b = LDS offset
            const unsigned gOff   = (unsigned)t * (CIN * 2) + cv * 16;
            asm volatile("global_load_async_to_lds_b128 %0, %1, %2"
                         :: "v"(ldsOff), "v"(gOff), "s"(actB) : "memory");
        } else {
            bf16x8 z;
#pragma unroll
            for (int i = 0; i < 8; ++i) z[i] = (bf16_t)0.0f;
            *(bf16x8*)dstp = z;            // zero halo / tail rows
        }
    }
    asm volatile("s_wait_asynccnt 0x0" ::: "memory");
    __syncthreads();

    if (o0 < Cout) {
        const bf16_t* wBase  = wT;
        const float*  shBase = shift;
        if (subj) {
            const int s = subj[b];
            wBase += (size_t)s * wSubjStride;
            if (shBase) shBase += (size_t)s * shSubjStride;
        }

        f32x8 acc0, acc1;
#pragma unroll
        for (int i = 0; i < 8; ++i) { acc0[i] = 0.0f; acc1[i] = 0.0f; }

#pragma unroll
        for (int k = 0; k < KTAPS; ++k) {
            // A: lane<16 -> K {0..7,16..23}; lane>=16 -> K {8..15,24..31}
            const bf16_t* arow = wBase + ((size_t)k * Cout + (o0 + n)) * CIN + hh * 8;
#pragma unroll
            for (int c0 = 0; c0 < CIN; c0 += 32) {
                union { bf16x16 v; bf16x8 h2[2]; } A, B0, B1;
                A.h2[0] = *(const bf16x8*)(arow + c0);
                A.h2[1] = *(const bf16x8*)(arow + c0 + 16);
                // B from LDS: lanes 0-15 K {0..7,16..23}, 16-31 K {8..15,24..31}, N = lane&15
                const bf16_t* b0 = &sh[(n + k) * PITCH + hh * 8 + c0];
                const bf16_t* b1 = &sh[(16 + n + k) * PITCH + hh * 8 + c0];
                B0.h2[0] = *(const bf16x8*)(b0);
                B0.h2[1] = *(const bf16x8*)(b0 + 16);
                B1.h2[0] = *(const bf16x8*)(b1);
                B1.h2[1] = *(const bf16x8*)(b1 + 16);
                acc0 = __builtin_amdgcn_wmma_f32_16x16x32_bf16(
                    false, A.v, false, B0.v, (short)0, acc0, false, false);
                acc1 = __builtin_amdgcn_wmma_f32_16x16x32_bf16(
                    false, A.v, false, B1.v, (short)0, acc1, false, false);
            }
        }

        // Epilogue. D layout: element i -> row m = hh*8 + i, col n.
#pragma unroll
        for (int tt = 0; tt < 2; ++tt) {
            const int    t     = t0 + tt * 16 + n;
            const size_t obase = ((size_t)b * TP + t) * Cout + o0 + hh * 8;
            const f32x8& acc   = tt ? acc1 : acc0;

            float rv[8];
            if (resid) {
                bf16x8 r = *(const bf16x8*)(resid + obase);
#pragma unroll
                for (int i = 0; i < 8; ++i) rv[i] = (float)r[i];
            } else {
#pragma unroll
                for (int i = 0; i < 8; ++i) rv[i] = 0.0f;
            }

            bf16x8 ov;
#pragma unroll
            for (int i = 0; i < 8; ++i) {
                const int oc = o0 + hh * 8 + i;
                float v = acc[i];
                if (scale)  v *= scale[oc];
                if (shBase) v += shBase[oc];
                v += rv[i];
                if (doGelu) v = gelu_exact(v);
                if (t >= TV) v = 0.0f;   // keep pad rows exactly zero
                ov[i] = (bf16_t)v;
            }
            *(bf16x8*)(out + obase) = ov;
        }
    }
}

// ---------------------------------------------------------------------------
// Head
// ---------------------------------------------------------------------------

// pooled[b][0:256] = mean_t act3[b][t][:]; pooled[b][256:272] = emb[subj[b]]
__global__ void k_pool(const bf16_t* __restrict__ act, const float* __restrict__ emb,
                       const int* __restrict__ subj, float* __restrict__ pooled) {
    const int b = blockIdx.x;
    const int j = threadIdx.x;            // blockDim 288
    if (j >= H2 + NE) return;
    float v;
    if (j < H2) {
        float s = 0.0f;
        const bf16_t* p = act + (size_t)b * TP * H2 + j;
        for (int t = 0; t < TV; ++t) s += (float)p[(size_t)t * H2];
        v = s * (1.0f / (float)TV);
    } else {
        v = emb[subj[b] * NE + (j - H2)];
    }
    pooled[(size_t)b * (H2 + NE) + j] = v;
}

__global__ void k_head1(const float* __restrict__ pooled, const float* __restrict__ w,
                        const float* __restrict__ bias, float* __restrict__ h) {
    int idx = blockIdx.x * blockDim.x + threadIdx.x;
    if (idx >= NB * H1) return;
    int o = idx % H1, b = idx / H1;
    const int J = H2 + NE;
    float s = bias[o];
    for (int j = 0; j < J; ++j) s += pooled[(size_t)b * J + j] * w[(size_t)o * J + j];
    h[idx] = fmaxf(s, 0.0f);
}

__global__ void k_head2(const float* __restrict__ h, const float* __restrict__ w,
                        const float* __restrict__ bias, float* __restrict__ out) {
    int idx = blockIdx.x * blockDim.x + threadIdx.x;
    if (idx >= NB * NCLS) return;
    int nc = idx % NCLS, b = idx / NCLS;
    float s = bias[nc];
    for (int o = 0; o < H1; ++o) s += h[(size_t)b * H1 + o] * w[(size_t)nc * H1 + o];
    out[idx] = s;
}

// ---------------------------------------------------------------------------
extern "C" void kernel_launch(void* const* d_in, const int* in_sizes, int n_in,
                              void* d_out, int out_size, void* d_ws, size_t ws_size,
                              hipStream_t stream) {
    (void)in_sizes; (void)n_in; (void)out_size; (void)ws_size;
    const float* X        = (const float*)d_in[0];
    const int*   subj     = (const int*)  d_in[1];
    const float* attention= (const float*)d_in[2];
    const float* sa_w     = (const float*)d_in[3];
    const float* sa_b     = (const float*)d_in[4];
    const float* subj_w   = (const float*)d_in[5];
    const float* subj_b   = (const float*)d_in[6];
    const float* b1_c1w   = (const float*)d_in[7];
    const float* b1_c1b   = (const float*)d_in[8];
    const float* b1_c2w   = (const float*)d_in[9];
    const float* b1_c2b   = (const float*)d_in[10];
    const float* b1_g1    = (const float*)d_in[11];
    const float* b1_be1   = (const float*)d_in[12];
    const float* b1_g2    = (const float*)d_in[13];
    const float* b1_be2   = (const float*)d_in[14];
    const float* b2_c1w   = (const float*)d_in[15];
    const float* b2_c1b   = (const float*)d_in[16];
    const float* b2_c2w   = (const float*)d_in[17];
    const float* b2_c2b   = (const float*)d_in[18];
    const float* b2_g1    = (const float*)d_in[19];
    const float* b2_be1   = (const float*)d_in[20];
    const float* b2_g2    = (const float*)d_in[21];
    const float* b2_be2   = (const float*)d_in[22];
    const float* b2_skw   = (const float*)d_in[23];
    const float* b2_skb   = (const float*)d_in[24];
    const float* b3_c1w   = (const float*)d_in[25];
    const float* b3_c1b   = (const float*)d_in[26];
    const float* b3_c2w   = (const float*)d_in[27];
    const float* b3_c2b   = (const float*)d_in[28];
    const float* b3_g1    = (const float*)d_in[29];
    const float* b3_be1   = (const float*)d_in[30];
    const float* b3_g2    = (const float*)d_in[31];
    const float* b3_be2   = (const float*)d_in[32];
    const float* emb      = (const float*)d_in[33];
    const float* head_w1  = (const float*)d_in[34];
    const float* head_b1  = (const float*)d_in[35];
    const float* head_w2  = (const float*)d_in[36];
    const float* head_b2  = (const float*)d_in[37];
    float* out = (float*)d_out;

    // --- carve workspace ---
    char* ws = (char*)d_ws;
    size_t off = 0;
    auto carve = [&](size_t bytes) -> void* {
        void* p = ws + off;
        off += (bytes + 255) & ~(size_t)255;
        return p;
    };
    float*  attn   = (float*) carve((size_t)CIN0 * CIN0 * 4);
    float*  W1     = (float*) carve((size_t)H1 * CIN0 * 4);
    bf16_t* Mw     = (bf16_t*)carve((size_t)4 * H1 * CX * 2);
    float*  Mb     = (float*) carve((size_t)4 * H1 * 4);
    bf16_t* wb1c1  = (bf16_t*)carve((size_t)3 * H1 * H1 * 2);
    bf16_t* wb1c2  = (bf16_t*)carve((size_t)3 * H1 * H1 * 2);
    bf16_t* wb2c1  = (bf16_t*)carve((size_t)3 * H2 * H1 * 2);
    bf16_t* wb2c2  = (bf16_t*)carve((size_t)3 * H2 * H2 * 2);
    bf16_t* wb2sk  = (bf16_t*)carve((size_t)H2 * H1 * 2);
    bf16_t* wb3c1  = (bf16_t*)carve((size_t)3 * H2 * H2 * 2);
    bf16_t* wb3c2  = (bf16_t*)carve((size_t)3 * H2 * H2 * 2);
    float* sc11 = (float*)carve(H1*4); float* sh11 = (float*)carve(H1*4);
    float* sc12 = (float*)carve(H1*4); float* sh12 = (float*)carve(H1*4);
    float* sc21 = (float*)carve(H2*4); float* sh21 = (float*)carve(H2*4);
    float* sc22 = (float*)carve(H2*4); float* sh22 = (float*)carve(H2*4);
    float* sc31 = (float*)carve(H2*4); float* sh31 = (float*)carve(H2*4);
    float* sc32 = (float*)carve(H2*4); float* sh32 = (float*)carve(H2*4);
    float*  pooled = (float*) carve((size_t)NB * (H2 + NE) * 4);
    float*  hmid   = (float*) carve((size_t)NB * H1 * 4);
    bf16_t* Xt     = (bf16_t*)carve((size_t)NB * TP * CX * 2);
    const size_t actBytes = (size_t)NB * TP * H2 * 2;   // 151 MB, fits 128ch too
    bf16_t* bufP = (bf16_t*)carve(actBytes);
    bf16_t* bufQ = (bf16_t*)carve(actBytes);
    bf16_t* bufR = (bf16_t*)carve(actBytes);

    // --- front-end fusion precompute ---
    k_softmax<<<CIN0, 128, 0, stream>>>(attention, attn);
    k_w1<<<(H1 * CIN0 + 255) / 256, 256, 0, stream>>>(sa_w, attn, W1);
    k_msub<<<(4 * H1 * CX + 255) / 256, 256, 0, stream>>>(subj_w, W1, Mw);
    k_mb<<<(4 * H1 + 255) / 256, 256, 0, stream>>>(subj_w, sa_b, subj_b, Mb);
    k_convert_x<<<dim3(TP, NB), 288, 0, stream>>>(X, Xt);

    // --- weight / BN prep ---
    k_wprep<<<(3*H1*H1 + 255)/256, 256, 0, stream>>>(b1_c1w, wb1c1, H1, H1, 3);
    k_wprep<<<(3*H1*H1 + 255)/256, 256, 0, stream>>>(b1_c2w, wb1c2, H1, H1, 3);
    k_wprep<<<(3*H2*H1 + 255)/256, 256, 0, stream>>>(b2_c1w, wb2c1, H2, H1, 3);
    k_wprep<<<(3*H2*H2 + 255)/256, 256, 0, stream>>>(b2_c2w, wb2c2, H2, H2, 3);
    k_wprep<<<(H2*H1 + 255)/256,   256, 0, stream>>>(b2_skw, wb2sk, H2, H1, 1);
    k_wprep<<<(3*H2*H2 + 255)/256, 256, 0, stream>>>(b3_c1w, wb3c1, H2, H2, 3);
    k_wprep<<<(3*H2*H2 + 255)/256, 256, 0, stream>>>(b3_c2w, wb3c2, H2, H2, 3);
    k_bnprep<<<1, 256, 0, stream>>>(b1_c1b, b1_g1, b1_be1, sc11, sh11, H1);
    k_bnprep<<<1, 256, 0, stream>>>(b1_c2b, b1_g2, b1_be2, sc12, sh12, H1);
    k_bnprep<<<1, 256, 0, stream>>>(b2_c1b, b2_g1, b2_be1, sc21, sh21, H2);
    k_bnprep<<<1, 256, 0, stream>>>(b2_c2b, b2_g2, b2_be2, sc22, sh22, H2);
    k_bnprep<<<1, 256, 0, stream>>>(b3_c1b, b3_g1, b3_be1, sc31, sh31, H2);
    k_bnprep<<<1, 256, 0, stream>>>(b3_c2b, b3_g2, b3_be2, sc32, sh32, H2);

    const dim3 blk(128);
    const dim3 g128(TP / 32, H1 / 64, NB);   // (9, 2, 1024)
    const dim3 g256(TP / 32, H2 / 64, NB);   // (9, 4, 1024)

    // Stem: per-subject fused matrix [128 x 288] @ Xt[b] -> bufP
    conv_wmma<1, CX><<<g128, blk, 0, stream>>>(Xt, Mw, H1, nullptr, Mb,
                                               nullptr, bufP, 0,
                                               subj, (long)H1 * CX, (long)H1);
    // Block1 (128->128, identity skip)
    conv_wmma<3, H1><<<g128, blk, 0, stream>>>(bufP, wb1c1, H1, sc11, sh11,
                                               nullptr, bufQ, 1, nullptr, 0, 0);
    conv_wmma<3, H1><<<g128, blk, 0, stream>>>(bufQ, wb1c2, H1, sc12, sh12,
                                               bufP, bufR, 1, nullptr, 0, 0);
    // Block2 (128->256, 1x1 conv skip)
    conv_wmma<1, H1><<<g256, blk, 0, stream>>>(bufR, wb2sk, H2, nullptr, b2_skb,
                                               nullptr, bufP, 0, nullptr, 0, 0);
    conv_wmma<3, H1><<<g256, blk, 0, stream>>>(bufR, wb2c1, H2, sc21, sh21,
                                               nullptr, bufQ, 1, nullptr, 0, 0);
    conv_wmma<3, H2><<<g256, blk, 0, stream>>>(bufQ, wb2c2, H2, sc22, sh22,
                                               bufP, bufR, 1, nullptr, 0, 0);
    // Block3 (256->256, identity skip)
    conv_wmma<3, H2><<<g256, blk, 0, stream>>>(bufR, wb3c1, H2, sc31, sh31,
                                               nullptr, bufQ, 1, nullptr, 0, 0);
    conv_wmma<3, H2><<<g256, blk, 0, stream>>>(bufQ, wb3c2, H2, sc32, sh32,
                                               bufR, bufP, 1, nullptr, 0, 0);

    // Head
    k_pool<<<NB, 288, 0, stream>>>(bufP, emb, subj, pooled);
    k_head1<<<(NB * H1 + 255) / 256, 256, 0, stream>>>(pooled, head_w1, head_b1, hmid);
    k_head2<<<(NB * NCLS + 255) / 256, 256, 0, stream>>>(hmid, head_w2, head_b2, out);
}
